// RecurrentLayerNet_60017872994406
// MI455X (gfx1250) — compile-verified, matrix operand order checked by reading, not run on hardware
//
#include <hip/hip_runtime.h>

typedef __attribute__((ext_vector_type(16))) __bf16 v16bf;
typedef __attribute__((ext_vector_type(8)))  float  v8f;

union FragBF { v16bf v; uint4 q[2]; };

#define B_SZ     1024
#define D_IN     784
#define K_PAD    800
#define NH       4096
#define N_OUT    10
#define N_LAYERS 24

// ---- CDNA5 async global->LDS path (guarded so the file compiles either way) ----
#if defined(__has_builtin)
#if __has_builtin(__builtin_amdgcn_global_load_async_to_lds_b128)
#define HAVE_ASYNC_LDS 1
#endif
#if __has_builtin(__builtin_amdgcn_s_wait_asynccnt)
#define HAVE_WAIT_ASYNC_BUILTIN 1
#endif
#endif
#ifndef HAVE_ASYNC_LDS
#define HAVE_ASYNC_LDS 0
#endif

#if HAVE_ASYNC_LDS
#if defined(HAVE_WAIT_ASYNC_BUILTIN)
#define WAIT_ASYNC(n) __builtin_amdgcn_s_wait_asynccnt(n)
#else
#define WAIT_ASYNC(n) asm volatile("s_wait_asynccnt %0" ::"i"(n) : "memory")
#endif
typedef __attribute__((address_space(1))) void GlobAS;
typedef __attribute__((address_space(3))) void LdsAS;
typedef int v4i __attribute__((vector_size(16)));
typedef __attribute__((address_space(1))) v4i GV4;   // builtin expects AS1 int4*
typedef __attribute__((address_space(3))) v4i LV4;   // builtin expects AS3 int4*
#else
#define WAIT_ASYNC(n)
#endif

// ---------------- conversion kernels ----------------

__global__ __launch_bounds__(256) void cvt_pad_kernel(
    const float* __restrict__ src, __bf16* __restrict__ dst,
    int rows, int srcK, int dstK) {
  long i = (long)blockIdx.x * blockDim.x + threadIdx.x;
  long total = (long)rows * dstK;
  if (i >= total) return;
  int r = (int)(i / dstK);
  int k = (int)(i % dstK);
  float v = (k < srcK) ? src[(long)r * srcK + k] : 0.0f;
  dst[i] = (__bf16)v;
}

__global__ __launch_bounds__(256) void cvt_scale_kernel(
    const float* __restrict__ src, const float* __restrict__ scale,
    __bf16* __restrict__ dst, long n, int K) {
  long i = (long)blockIdx.x * blockDim.x + threadIdx.x;
  if (i >= n) return;
  int k = (int)(i % K);
  dst[i] = (__bf16)(src[i] * scale[k]);
}

// ---------------- bf16 WMMA GEMM: C = A @ W^T + bias ----------------
// A: M x K bf16 (stride lda), W: N x K bf16 (stride ldw)
// pre_out (optional, f32, stride N) streamed with non-temporal stores; act_out = relu(C) bf16.
// M,N % 128 == 0, K % 32 == 0 (and K >= 64). 256 threads = 8 waves (4 over M x 2 over N);
// each wave owns a 32x64 sub-tile = 2x4 v_wmma_f32_16x16x32_bf16 accumulators.
// LDS tiles are double-buffered, staged with GLOBAL_LOAD_ASYNC_TO_LDS_B128 (ASYNCcnt),
// with the last K-step peeled so the steady-state loop body is branchless.

__global__ __launch_bounds__(256) void gemm_bf16_relu_kernel(
    const __bf16* __restrict__ A, int lda,
    const __bf16* __restrict__ W, int ldw,
    const float* __restrict__ bias,
    float* __restrict__ pre_out,
    __bf16* __restrict__ act_out,
    int M, int N, int K) {
  __shared__ __bf16 As[2][128][40];   // +8 pad, row stride 80B (16B aligned)
  __shared__ __bf16 Ws[2][128][40];

  const int tid  = threadIdx.x;
  const int lane = tid & 31;
  const int wave = tid >> 5;
  const int wm   = wave & 3;   // wave row (4 x 32 = 128)
  const int wn   = wave >> 2;  // wave col (2 x 64 = 128)
  const int half = lane >> 4;  // 0: lanes 0-15, 1: lanes 16-31
  const int l16  = lane & 15;

  const int bm = blockIdx.y * 128;
  const int bn = blockIdx.x * 128;

  v8f acc[2][4];
#pragma unroll
  for (int mi = 0; mi < 2; ++mi)
#pragma unroll
    for (int ni = 0; ni < 4; ++ni) {
      v8f z = {0.f, 0.f, 0.f, 0.f, 0.f, 0.f, 0.f, 0.f};
      acc[mi][ni] = z;
    }

  const int ldRow = tid >> 2;        // 0..63 (two row passes cover 128)
  const int ldCol = (tid & 3) * 8;   // 0,8,16,24 (8 bf16 = 16B per thread)

  const int nsteps = K >> 5;

  // --- stage one 128x32 K-tile (A and W halves) into LDS buffer `buf` ---
  auto stage = [&](int step, int buf) {
    int kk = step << 5;
#pragma unroll
    for (int p = 0; p < 2; ++p) {
      int r = ldRow + p * 64;
      const __bf16* ga = &A[(long)(bm + r) * lda + kk + ldCol];
      const __bf16* gw = &W[(long)(bn + r) * ldw + kk + ldCol];
#if HAVE_ASYNC_LDS
      __builtin_amdgcn_global_load_async_to_lds_b128(
          (GV4*)(GlobAS*)ga, (LV4*)(LdsAS*)&As[buf][r][ldCol], 0, 0);
      __builtin_amdgcn_global_load_async_to_lds_b128(
          (GV4*)(GlobAS*)gw, (LV4*)(LdsAS*)&Ws[buf][r][ldCol], 0, 0);
#else
      *(uint4*)&As[buf][r][ldCol] = *(const uint4*)ga;
      *(uint4*)&Ws[buf][r][ldCol] = *(const uint4*)gw;
#endif
    }
  };

  // --- one K-step of WMMAs out of LDS buffer `cur` ---
  auto compute = [&](int cur) {
    // A fragments: 16x32 bf16 ISA layout -> half selects K groups {0-7,16-23} vs {8-15,24-31}
    FragBF af[2], wf[4];
#pragma unroll
    for (int mi = 0; mi < 2; ++mi) {
      int r = wm * 32 + mi * 16 + l16;
      af[mi].q[0] = *(const uint4*)&As[cur][r][half * 8];
      af[mi].q[1] = *(const uint4*)&As[cur][r][16 + half * 8];
    }
#pragma unroll
    for (int ni = 0; ni < 4; ++ni) {
      int r = wn * 64 + ni * 16 + l16;
      wf[ni].q[0] = *(const uint4*)&Ws[cur][r][half * 8];
      wf[ni].q[1] = *(const uint4*)&Ws[cur][r][16 + half * 8];
    }
#pragma unroll
    for (int mi = 0; mi < 2; ++mi)
#pragma unroll
      for (int ni = 0; ni < 4; ++ni)
        acc[mi][ni] = __builtin_amdgcn_wmma_f32_16x16x32_bf16(
            /*neg_a=*/false, af[mi].v, /*neg_b=*/false, wf[ni].v,
            /*c_mod=*/(short)0, acc[mi][ni],
            /*reuse_a=*/false, /*reuse_b=*/false);
  };

  stage(0, 0);

  // Steady state: branchless body (stage next, wait current's 4 async loads, barrier, WMMA).
  for (int s = 0; s < nsteps - 1; ++s) {
    int cur = s & 1;
    stage(s + 1, cur ^ 1);
    WAIT_ASYNC(4);
    __syncthreads();
    compute(cur);
    __syncthreads();   // guard LDS buffer reuse two steps ahead
  }

  // Peeled last step: drain all async loads.
  WAIT_ASYNC(0);
  __syncthreads();
  compute((nsteps - 1) & 1);

  // Epilogue: C layout: VGPR v -> row = v + 8*half; lanes carry N = l16.
  // pre_out is a 420 MB write-once stream -> non-temporal so W_sp/acts stay hot in L2.
#pragma unroll
  for (int ni = 0; ni < 4; ++ni) {
    int col = bn + wn * 64 + ni * 16 + l16;
    float b = bias[col];
#pragma unroll
    for (int mi = 0; mi < 2; ++mi) {
      int rowBase = bm + wm * 32 + mi * 16 + half * 8;
#pragma unroll
      for (int v = 0; v < 8; ++v) {
        float val = acc[mi][ni][v] + b;
        long idx = (long)(rowBase + v) * N + col;
        if (pre_out) __builtin_nontemporal_store(val, &pre_out[idx]);
        act_out[idx] = (__bf16)(val > 0.f ? val : 0.f);
      }
    }
  }
}

// ---------------- final 4096 -> 10 heads (tiny, VALU) ----------------

__global__ __launch_bounds__(256) void final_head_kernel(
    const __bf16* __restrict__ actA, const __bf16* __restrict__ actB,
    const float* __restrict__ Wf, const float* __restrict__ bf,
    const float* __restrict__ of,
    float* __restrict__ outA, float* __restrict__ outB) {
  int row = blockIdx.x;
  const __bf16* act = blockIdx.y ? actB : actA;
  float* out = blockIdx.y ? outB : outA;
  int tid = threadIdx.x;

  float partial[N_OUT];
#pragma unroll
  for (int j = 0; j < N_OUT; ++j) partial[j] = 0.f;

  for (int k = tid; k < NH; k += 256) {
    float a = (float)act[(long)row * NH + k] * of[k];
#pragma unroll
    for (int j = 0; j < N_OUT; ++j)
      partial[j] += a * Wf[(long)j * NH + k];
  }

  __shared__ float red[N_OUT];
  if (tid < N_OUT) red[tid] = 0.f;
  __syncthreads();
#pragma unroll
  for (int j = 0; j < N_OUT; ++j) atomicAdd(&red[j], partial[j]);
  __syncthreads();
  if (tid < N_OUT) out[(long)row * N_OUT + tid] = red[tid] + bf[tid];
}

// ---------------- launch ----------------

extern "C" void kernel_launch(void* const* d_in, const int* in_sizes, int n_in,
                              void* d_out, int out_size, void* d_ws, size_t ws_size,
                              hipStream_t stream) {
  (void)in_sizes; (void)n_in; (void)out_size; (void)ws_size;

  const float* x      = (const float*)d_in[0];
  const float* W_in   = (const float*)d_in[1];
  const float* b_in   = (const float*)d_in[2];
  const float* key_sp = (const float*)d_in[3];
  const float* W_sp   = (const float*)d_in[4];
  const float* b_sp   = (const float*)d_in[5];
  const float* key_f  = (const float*)d_in[6];
  const float* W_f    = (const float*)d_in[7];
  const float* b_f    = (const float*)d_in[8];
  // d_in[9] = time = 0; key_sp/key_f have a single row, so it is always row 0.

  float* out  = (float*)d_out;
  float* outA = out;
  float* outB = out + (size_t)B_SZ * N_OUT;
  float* pre  = out + (size_t)2 * B_SZ * N_OUT;  // 25 x 1024 x 4096 f32

  char* ws = (char*)d_ws;
  __bf16* xb   = (__bf16*)ws;  ws += (size_t)B_SZ * K_PAD * sizeof(__bf16);
  __bf16* Winb = (__bf16*)ws;  ws += (size_t)2 * NH * K_PAD * sizeof(__bf16);
  __bf16* Wspb = (__bf16*)ws;  ws += (size_t)NH * NH * sizeof(__bf16);
  __bf16* actA[2];
  __bf16* actB[2];
  actA[0] = (__bf16*)ws;  ws += (size_t)B_SZ * NH * sizeof(__bf16);
  actA[1] = (__bf16*)ws;  ws += (size_t)B_SZ * NH * sizeof(__bf16);
  actB[0] = (__bf16*)ws;  ws += (size_t)B_SZ * NH * sizeof(__bf16);
  actB[1] = (__bf16*)ws;  ws += (size_t)B_SZ * NH * sizeof(__bf16);

  // One-shot conversions: fold the +/-1 sign vector into W_sp; pad K 784 -> 800.
  {
    long n = (long)B_SZ * K_PAD;
    cvt_pad_kernel<<<(unsigned)((n + 255) / 256), 256, 0, stream>>>(x, xb, B_SZ, D_IN, K_PAD);
  }
  {
    long n = (long)2 * NH * K_PAD;
    cvt_pad_kernel<<<(unsigned)((n + 255) / 256), 256, 0, stream>>>(W_in, Winb, 2 * NH, D_IN, K_PAD);
  }
  {
    long n = (long)NH * NH;
    cvt_scale_kernel<<<(unsigned)((n + 255) / 256), 256, 0, stream>>>(W_sp, key_sp, Wspb, n, NH);
  }

  dim3 blk(256);
  dim3 grd(NH / 128, B_SZ / 128);

  // Input layer: r = x @ W_in^T + b_in. First half -> stream a (store pre0), second -> stream b.
  gemm_bf16_relu_kernel<<<grd, blk, 0, stream>>>(
      xb, K_PAD, Winb, K_PAD, b_in, pre /* slot 0 */, actA[0], B_SZ, NH, K_PAD);
  gemm_bf16_relu_kernel<<<grd, blk, 0, stream>>>(
      xb, K_PAD, Winb + (size_t)NH * K_PAD, K_PAD, b_in + NH, nullptr, actB[0], B_SZ, NH, K_PAD);

  // 24 recurrent layers; W_sp (bf16, 33 MB) stays resident in L2.
  int cur = 0;
  for (int l = 0; l < N_LAYERS; ++l) {
    int nxt = cur ^ 1;
    gemm_bf16_relu_kernel<<<grd, blk, 0, stream>>>(
        actA[cur], NH, Wspb, NH, b_sp,
        pre + (size_t)(l + 1) * B_SZ * NH, actA[nxt], B_SZ, NH, NH);
    gemm_bf16_relu_kernel<<<grd, blk, 0, stream>>>(
        actB[cur], NH, Wspb, NH, b_sp,
        nullptr, actB[nxt], B_SZ, NH, NH);
    cur = nxt;
  }

  // Heads: 1024 x 10 per stream (sign vector key_f applied inline).
  dim3 fgrd(B_SZ, 2);
  final_head_kernel<<<fgrd, 256, 0, stream>>>(actA[cur], actB[cur], W_f, b_f, key_f, outA, outB);
}
